// LoRA_qkv_47656957116918
// MI455X (gfx1250) — matrix operand order repaired and checked by reading, not compiled, not run on hardware
//
#include <hip/hip_runtime.h>
#include <hip/hip_bf16.h>

typedef __attribute__((ext_vector_type(16))) __bf16 v16bf;
typedef __attribute__((ext_vector_type(8)))  __bf16 v8bf;
typedef __attribute__((ext_vector_type(8)))  float  v8f;

#define D_DIM 1024
#define N_DIM 3072
#define R_DIM 16
#define BLK_M 128
#define BLK_N 128
#define BLK_K 32

// ---------------- Kernel 1: split x (fp32) into bf16 hi/lo ----------------
__global__ void split_x_kernel(const float* __restrict__ x,
                               __bf16* __restrict__ xh,
                               __bf16* __restrict__ xl, int n) {
    int i = blockIdx.x * blockDim.x + threadIdx.x;
    if (i < n) {
        float v = x[i];
        __bf16 h = (__bf16)v;
        __bf16 l = (__bf16)(v - (float)h);
        xh[i] = h;
        xl[i] = l;
    }
}

// ---- Kernel 2: fold rank-16 LoRA into W_qkv, split into bf16 hi/lo -------
__global__ void build_w_kernel(const float* __restrict__ Wqkv,
                               const float* __restrict__ Waq,
                               const float* __restrict__ Wbq,
                               const float* __restrict__ Wav,
                               const float* __restrict__ Wbv,
                               __bf16* __restrict__ wh,
                               __bf16* __restrict__ wl) {
    int idx = blockIdx.x * blockDim.x + threadIdx.x;   // over N_DIM*D_DIM
    int e = idx >> 10;
    int d = idx & 1023;
    float acc = Wqkv[idx];
    if (e < D_DIM) {
        const float* wb = Wbq + e * R_DIM;
#pragma unroll
        for (int r = 0; r < R_DIM; ++r)
            acc += wb[r] * Waq[r * D_DIM + d];
    } else if (e >= 2 * D_DIM) {
        const float* wb = Wbv + (e - 2 * D_DIM) * R_DIM;
#pragma unroll
        for (int r = 0; r < R_DIM; ++r)
            acc += wb[r] * Wav[r * D_DIM + d];
    }
    __bf16 h = (__bf16)acc;
    wh[idx] = h;
    wl[idx] = (__bf16)(acc - (float)h);
}

// ---------------- async global -> LDS copy (CDNA5, ASYNCcnt) --------------
__device__ __forceinline__ void async_ld16(unsigned lds_addr, unsigned voff_bytes,
                                           const void* base) {
    // GVS addressing: mem = SADDR64 + VADDR32 ; LDS dest = VGPR lds addr
    asm volatile("global_load_async_to_lds_b128 %0, %1, %2"
                 :: "v"(lds_addr), "v"(voff_bytes), "s"(base)
                 : "memory");
}

__device__ __forceinline__ unsigned lds32(const void* p) {
    return (unsigned)(unsigned long long)p;   // low 32 bits = LDS byte address
}

// ---------------- Kernel 3: split-bf16 WMMA GEMM + bias -------------------
// out[M, N] = x@W_eff^T (hi*hi + hi*lo + lo*hi) + bias, fp32 accumulate.
__global__ __launch_bounds__(256) void
lora_qkv_gemm_kernel(const __bf16* __restrict__ xh, const __bf16* __restrict__ xl,
                     const __bf16* __restrict__ wh, const __bf16* __restrict__ wl,
                     const float* __restrict__ bias, float* __restrict__ out) {
    // double-buffered: [buf][array: Ah,Al,Bh,Bl][128*32 bf16] = 64 KB total
    __shared__ __align__(16) __bf16 smem[2][4][BLK_M * BLK_K];

    const int tid  = threadIdx.x;
    const int lane = tid & 31;          // wave32
    const int wave = tid >> 5;          // 8 waves
    const int wm   = wave & 3;          // 32-row strip within 128
    const int wn   = wave >> 2;         // 64-col strip within 128
    const int Mbase = blockIdx.y * BLK_M;
    const int Nbase = blockIdx.x * BLK_N;

    const int lhalf = (lane < 16) ? 0 : 1;
    const int l15   = lane & 15;

    v8f acc[2][4] = {};   // 2 M-tiles x 4 N-tiles of 16x16 fp32

    // per-thread 16B chunks of the 128x32 tile: 512 chunks / 256 threads = 2
    const int row0 = (tid * 2) >> 2;
    const int kc0  = ((tid * 2) & 3) * 8;
    const int row1 = (tid * 2 + 1) >> 2;
    const int kc1  = ((tid * 2 + 1) & 3) * 8;
    const int s0   = row0 * BLK_K + kc0;      // LDS element offset
    const int s1   = row1 * BLK_K + kc1;

    // global byte offsets (GVS 32-bit voffset), advance by BLK_K*2 = 64 B/step
    unsigned vA0 = (unsigned)((Mbase + row0) * D_DIM + kc0) * 2u;
    unsigned vA1 = (unsigned)((Mbase + row1) * D_DIM + kc1) * 2u;
    unsigned vB0 = (unsigned)((Nbase + row0) * D_DIM + kc0) * 2u;
    unsigned vB1 = (unsigned)((Nbase + row1) * D_DIM + kc1) * 2u;

    // prologue: async-fill buffer 0 with K-step 0
    async_ld16(lds32(&smem[0][0][s0]), vA0, xh);
    async_ld16(lds32(&smem[0][0][s1]), vA1, xh);
    async_ld16(lds32(&smem[0][1][s0]), vA0, xl);
    async_ld16(lds32(&smem[0][1][s1]), vA1, xl);
    async_ld16(lds32(&smem[0][2][s0]), vB0, wh);
    async_ld16(lds32(&smem[0][2][s1]), vB1, wh);
    async_ld16(lds32(&smem[0][3][s0]), vB0, wl);
    async_ld16(lds32(&smem[0][3][s1]), vB1, wl);
    asm volatile("s_wait_asynccnt 0x0" ::: "memory");
    __syncthreads();

    const int KSTEPS = D_DIM / BLK_K;   // 32
    for (int ks = 0; ks < KSTEPS; ++ks) {
        const int cur = ks & 1;
        // issue async copy of next K-step into the other buffer
        if (ks + 1 < KSTEPS) {
            const int nxt = cur ^ 1;
            vA0 += BLK_K * 2; vA1 += BLK_K * 2;
            vB0 += BLK_K * 2; vB1 += BLK_K * 2;
            async_ld16(lds32(&smem[nxt][0][s0]), vA0, xh);
            async_ld16(lds32(&smem[nxt][0][s1]), vA1, xh);
            async_ld16(lds32(&smem[nxt][1][s0]), vA0, xl);
            async_ld16(lds32(&smem[nxt][1][s1]), vA1, xl);
            async_ld16(lds32(&smem[nxt][2][s0]), vB0, wh);
            async_ld16(lds32(&smem[nxt][2][s1]), vB1, wh);
            async_ld16(lds32(&smem[nxt][3][s0]), vB0, wl);
            async_ld16(lds32(&smem[nxt][3][s1]), vB1, wl);
        }

        const __bf16* sAh = smem[cur][0];
        const __bf16* sAl = smem[cur][1];
        const __bf16* sBh = smem[cur][2];
        const __bf16* sBl = smem[cur][3];

        // ---- A fragments (16x32 bf16): lanes 0-15 hold K {0..7,16..23},
        //      lanes 16-31 hold K {8..15,24..31}; row = lane&15.
        v16bf aH[2], aL[2];
#pragma unroll
        for (int i = 0; i < 2; ++i) {
            int rowA = wm * 32 + i * 16 + l15;
            int k0   = lhalf * 8;
            v8bf h0 = *(const v8bf*)(sAh + rowA * BLK_K + k0);
            v8bf h1 = *(const v8bf*)(sAh + rowA * BLK_K + k0 + 16);
            aH[i] = __builtin_shufflevector(h0, h1, 0,1,2,3,4,5,6,7,8,9,10,11,12,13,14,15);
            v8bf l0 = *(const v8bf*)(sAl + rowA * BLK_K + k0);
            v8bf l1 = *(const v8bf*)(sAl + rowA * BLK_K + k0 + 16);
            aL[i] = __builtin_shufflevector(l0, l1, 0,1,2,3,4,5,6,7,8,9,10,11,12,13,14,15);
        }

        // ---- B fragments (32x16 bf16): lanes 0-15 hold K {0..7,16..23},
        //      lanes 16-31 hold K {8..15,24..31}; column = lane&15.
#pragma unroll
        for (int j = 0; j < 4; ++j) {
            int colB = wn * 64 + j * 16 + l15;
            int kb   = lhalf * 8;
            v8bf h0 = *(const v8bf*)(sBh + colB * BLK_K + kb);
            v8bf h1 = *(const v8bf*)(sBh + colB * BLK_K + kb + 16);
            v16bf bH = __builtin_shufflevector(h0, h1, 0,1,2,3,4,5,6,7,8,9,10,11,12,13,14,15);
            v8bf l0 = *(const v8bf*)(sBl + colB * BLK_K + kb);
            v8bf l1 = *(const v8bf*)(sBl + colB * BLK_K + kb + 16);
            v16bf bL = __builtin_shufflevector(l0, l1, 0,1,2,3,4,5,6,7,8,9,10,11,12,13,14,15);
#pragma unroll
            for (int i = 0; i < 2; ++i) {
                acc[i][j] = __builtin_amdgcn_wmma_f32_16x16x32_bf16(
                    false, aH[i], false, bH, (short)0, acc[i][j], false, false);
                acc[i][j] = __builtin_amdgcn_wmma_f32_16x16x32_bf16(
                    false, aH[i], false, bL, (short)0, acc[i][j], false, false);
                acc[i][j] = __builtin_amdgcn_wmma_f32_16x16x32_bf16(
                    false, aL[i], false, bH, (short)0, acc[i][j], false, false);
            }
        }

        // my next-buffer async loads must have landed before anyone reads them
        asm volatile("s_wait_asynccnt 0x0" ::: "memory");
        __syncthreads();
    }

    // ---- epilogue: D layout = VGPR r -> row r (lanes 0-15) / row 8+r
    //      (lanes 16-31); lane&15 -> column. Add bias, store fp32.
#pragma unroll
    for (int i = 0; i < 2; ++i) {
        int rowb = Mbase + wm * 32 + i * 16 + lhalf * 8;
#pragma unroll
        for (int j = 0; j < 4; ++j) {
            int col = Nbase + wn * 64 + j * 16 + l15;
            float bv = bias[col];
#pragma unroll
            for (int r = 0; r < 8; ++r) {
                out[(size_t)(rowb + r) * N_DIM + col] = acc[i][j][r] + bv;
            }
        }
    }
}

extern "C" void kernel_launch(void* const* d_in, const int* in_sizes, int n_in,
                              void* d_out, int out_size, void* d_ws, size_t ws_size,
                              hipStream_t stream) {
    const float* x    = (const float*)d_in[0];   // [4,4096,1024]
    const float* Wqkv = (const float*)d_in[1];   // [3072,1024]
    const float* bq   = (const float*)d_in[2];   // [3072]
    const float* Waq  = (const float*)d_in[3];   // [16,1024]
    const float* Wbq  = (const float*)d_in[4];   // [1024,16]
    const float* Wav  = (const float*)d_in[5];   // [16,1024]
    const float* Wbv  = (const float*)d_in[6];   // [1024,16]
    float* out = (float*)d_out;

    const int M  = in_sizes[0] / D_DIM;          // 16384
    const int MK = in_sizes[0];                  // 16777216
    const int NK = N_DIM * D_DIM;                // 3145728

    // workspace layout: xh, xl [M*K] bf16 ; wh, wl [N*K] bf16 (~76 MiB)
    __bf16* xh = (__bf16*)d_ws;
    __bf16* xl = xh + MK;
    __bf16* wh = xl + MK;
    __bf16* wl = wh + NK;

    split_x_kernel<<<(MK + 255) / 256, 256, 0, stream>>>(x, xh, xl, MK);
    build_w_kernel<<<(NK + 255) / 256, 256, 0, stream>>>(Wqkv, Waq, Wbq, Wav, Wbv, wh, wl);

    dim3 grid(N_DIM / BLK_N, M / BLK_M);         // (24, 128)
    lora_qkv_gemm_kernel<<<grid, 256, 0, stream>>>(xh, xl, wh, wl, bq, out);
}
